// KEANN_13417477833153
// MI455X (gfx1250) — compile-verified, matrix operand order checked by reading, not run on hardware
//
#include <hip/hip_runtime.h>

// ---------------------------------------------------------------------------
// Problem constants (from reference): B=32, L=128, E=256, H=256, C=4
// N = B*L = 4096 ; 4H = 1024 ; 2H = 512 ; out cols = 2L*2H = 131072
// ---------------------------------------------------------------------------
#define NROWS 4096
#define G4    1024
#define TWOH  512
#define OUTC  131072

typedef __attribute__((ext_vector_type(16))) _Float16 v16h;
typedef __attribute__((ext_vector_type(8)))  float    v8f;

// ---------------- workspace layout (bytes) ----------------
static constexpr size_t OFF_E16   = 0;                         // 8192*256 f16   = 4 MiB
static constexpr size_t OFF_WIH16 = OFF_E16   + (size_t)8192*256*2;      // 2*1024*256 f16
static constexpr size_t OFF_WHH16 = OFF_WIH16 + (size_t)2*1024*256*2;    // 2*1024*256 f16
static constexpr size_t OFF_RMT16 = OFF_WHH16 + (size_t)2*1024*256*2;    // 512*512 f16 (transposed)
static constexpr size_t OFF_XW16  = OFF_RMT16 + (size_t)512*512*2;       // 2*2*4096*1024 f16 = 32 MiB
static constexpr size_t OFF_R32   = OFF_XW16  + (size_t)4*4096*1024*2;   // 2*4096*512 f32 = 16 MiB
static constexpr size_t OFF_R16   = OFF_R32   + (size_t)2*4096*512*4;    // 2*4096*512 f16 = 8 MiB
static constexpr size_t OFF_T1R   = OFF_R16   + (size_t)2*4096*512*2;    // 4096*512 f16
static constexpr size_t OFF_EXPA  = OFF_T1R   + (size_t)4096*512*2;      // 4096*4096 f16 = 32 MiB
static constexpr size_t OFF_RS    = OFF_EXPA  + (size_t)4096*4096*2;     // 4096 f32
static constexpr size_t OFF_CS    = OFF_RS    + 4096*4;
static constexpr size_t OFF_SF1   = OFF_CS    + 4096*4;
static constexpr size_t OFF_SF2   = OFF_SF1   + 4096*4;
static constexpr size_t OFF_OUT   = OFF_SF2   + 4096*4;                  // 32*131072 f32 = 16 MiB
static constexpr size_t OFF_MU    = OFF_OUT   + (size_t)32*OUTC*4;
static constexpr size_t OFF_RSTD  = OFF_MU    + (size_t)OUTC*4;

// ---------------- WMMA helpers ----------------
__device__ __forceinline__ v8f wmma_f16(v16h a, v16h b, v8f c) {
  // D = A(16x32 f16) * B(32x16 f16) + C(16x16 f32)
  return __builtin_amdgcn_wmma_f32_16x16x32_f16(false, a, false, b, (short)0, c, false, false);
}

// Load a 16x32 f16 fragment for this lane.
// rowp: pointer to element k=0 (of the 32-wide K block) of this lane's row
// (row = M for A-fragments, row = N (i.e. row of W / of B^T) for B-fragments).
// ISA 7.12.2: lanes 0-15 K-pairs {0,1},{2,3},{4,5},{6,7},{16,17}..{22,23};
// lanes 16-31 add +8. All pairs are 4B-contiguous in row-major memory.
__device__ __forceinline__ v16h load_frag(const _Float16* rowp, int khalf) {
  union { v16h h; unsigned int u[8]; } x;
  const unsigned int* q = (const unsigned int*)rowp;
  #pragma unroll
  for (int v = 0; v < 8; ++v) {
    int kb = ((v < 4) ? 2 * v : 16 + 2 * (v - 4)) + 8 * khalf;
    x.u[v] = q[kb >> 1];
  }
  return x.h;
}

__device__ __forceinline__ float sigm(float x) { return 1.f / (1.f + __expf(-x)); }

// ---------------- kernels ----------------

// Embedding gather (padding row 0 -> 0), f32 table -> f16 activations.
__global__ void ke_embed(const int* __restrict__ x, const float* __restrict__ tbl,
                         _Float16* __restrict__ e16) {
  int idx = blockIdx.x * 256 + threadIdx.x;       // 2*4096*256 total
  int row = idx >> 8, col = idx & 255;
  int tok = x[row];
  float v = (tok == 0) ? 0.f : tbl[(size_t)tok * 256 + col];
  e16[idx] = (_Float16)v;
}

__global__ void ke_cvt(const float* __restrict__ s, _Float16* __restrict__ d, int n) {
  int i = blockIdx.x * 256 + threadIdx.x;
  if (i < n) d[i] = (_Float16)s[i];
}

// rand_matrix (512x512) -> transposed f16 so GEMM B-side reads rows contiguously.
__global__ void ke_cvtT(const float* __restrict__ s, _Float16* __restrict__ d) {
  int i = blockIdx.x * 256 + threadIdx.x;         // 512*512
  int r = i >> 9, c = i & 511;
  d[c * 512 + r] = (_Float16)s[i];
}

// xw = e(8192x256) * W_ih_dir^T (256x1024) + b_dir, stored f16 as [input][dir][4096][1024]
__global__ void ke_gemm_xw(const _Float16* __restrict__ A, const _Float16* __restrict__ Wall,
                           const float* __restrict__ bf, const float* __restrict__ bb,
                           _Float16* __restrict__ xw) {
  const int dir = blockIdx.z;
  const _Float16* Bm = Wall + (size_t)dir * 1024 * 256;
  const float* bias = dir ? bb : bf;
  const int wave = threadIdx.x >> 5, lane = threadIdx.x & 31;
  const int hi = lane >> 4, ln = lane & 15;
  const int mrow = (blockIdx.y * 8 + wave) * 16;
  const int nbase = blockIdx.x * 64;
  v8f acc[4] = {};
  const _Float16* arow = A + (size_t)(mrow + ln) * 256;
  for (int k0 = 0; k0 < 256; k0 += 32) {
    v16h a = load_frag(arow + k0, hi);
    #pragma unroll
    for (int nt = 0; nt < 4; ++nt) {
      v16h b = load_frag(Bm + (size_t)(nbase + nt * 16 + ln) * 256 + k0, hi);
      acc[nt] = wmma_f16(a, b, acc[nt]);
    }
  }
  #pragma unroll
  for (int nt = 0; nt < 4; ++nt) {
    int n = nbase + nt * 16 + ln;
    float bv = bias[n];
    #pragma unroll
    for (int r = 0; r < 8; ++r) {
      int row = mrow + hi * 8 + r;
      int input = row >> 12, rin = row & 4095;
      xw[(((size_t)input * 2 + dir) * 4096 + rin) * 1024 + n] = (_Float16)(acc[nt][r] + bv);
    }
  }
}

// One workgroup per (input,dir). 512 threads = 16 waves.
// Wave w owns gate-tiles {w, w+16, w+32, w+48} x both M-tiles, so the i/f/g/o
// pre-activations of each (batch,hidden) element sit in the same lane slot.
// h is f16 in LDS (A-side of WMMA), c stays in accumulator-layout registers.
__global__ void ke_lstm(const _Float16* __restrict__ xw, const _Float16* __restrict__ Whh,
                        float* __restrict__ r32, _Float16* __restrict__ r16) {
  const int input = blockIdx.x >> 1, dir = blockIdx.x & 1;
  const _Float16* whh = Whh + (size_t)dir * 1024 * 256;
  const _Float16* xwb = xw + ((size_t)input * 2 + dir) * 4096 * 1024;
  __shared__ _Float16 hbuf[32 * 256];               // 16 KiB
  const int tid = threadIdx.x;
  const int wv = tid >> 5, lane = tid & 31;
  const int hi = lane >> 4, ln = lane & 15;
  for (int i = tid; i < 32 * 256; i += 512) hbuf[i] = (_Float16)0.f;
  v8f c[2] = {};
  __syncthreads();
  for (int t = 0; t < 128; ++t) {
    v8f acc[2][4] = {};
    for (int k0 = 0; k0 < 256; k0 += 32) {
      v16h a0 = load_frag(&hbuf[(0 + ln) * 256 + k0], hi);
      v16h a1 = load_frag(&hbuf[(16 + ln) * 256 + k0], hi);
      #pragma unroll
      for (int g = 0; g < 4; ++g) {
        v16h b = load_frag(whh + (size_t)((wv + 16 * g) * 16 + ln) * 256 + k0, hi);
        acc[0][g] = wmma_f16(a0, b, acc[0][g]);
        acc[1][g] = wmma_f16(a1, b, acc[1][g]);
      }
    }
    __syncthreads();                                 // everyone done reading h
    const int l = dir ? (127 - t) : t;
    #pragma unroll
    for (int mt = 0; mt < 2; ++mt) {
      #pragma unroll
      for (int r = 0; r < 8; ++r) {
        int m = mt * 16 + hi * 8 + r;                // batch index
        int hc = wv * 16 + ln;                       // hidden column
        const _Float16* xp = xwb + ((size_t)m * 128 + l) * 1024;
        float pi = acc[mt][0][r] + (float)xp[hc];
        float pf = acc[mt][1][r] + (float)xp[256 + hc];
        float pg = acc[mt][2][r] + (float)xp[512 + hc];
        float po = acc[mt][3][r] + (float)xp[768 + hc];
        float cn = sigm(pf) * c[mt][r] + sigm(pi) * tanhf(pg);
        c[mt][r] = cn;
        float h = sigm(po) * tanhf(cn);
        hbuf[m * 256 + hc] = (_Float16)h;
        size_t o = ((size_t)input * 4096 + (size_t)m * 128 + l) * 512 + dir * 256 + hc;
        r32[o] = h;
        r16[o] = (_Float16)h;
      }
    }
    __syncthreads();
  }
}

// t1R = t1(4096x512) * R(512x512), R pre-transposed so B rows are contiguous.
__global__ void ke_gemm_t1R(const _Float16* __restrict__ A, const _Float16* __restrict__ Bm,
                            _Float16* __restrict__ D) {
  const int wave = threadIdx.x >> 5, lane = threadIdx.x & 31;
  const int hi = lane >> 4, ln = lane & 15;
  const int mrow = (blockIdx.y * 8 + wave) * 16;
  const int nbase = blockIdx.x * 64;
  v8f acc[4] = {};
  const _Float16* arow = A + (size_t)(mrow + ln) * 512;
  for (int k0 = 0; k0 < 512; k0 += 32) {
    v16h a = load_frag(arow + k0, hi);
    #pragma unroll
    for (int nt = 0; nt < 4; ++nt) {
      v16h b = load_frag(Bm + (size_t)(nbase + nt * 16 + ln) * 512 + k0, hi);
      acc[nt] = wmma_f16(a, b, acc[nt]);
    }
  }
  #pragma unroll
  for (int nt = 0; nt < 4; ++nt) {
    int n = nbase + nt * 16 + ln;
    #pragma unroll
    for (int r = 0; r < 8; ++r) {
      int row = mrow + hi * 8 + r;
      D[(size_t)row * 512 + n] = (_Float16)acc[nt][r];
    }
  }
}

// expA = exp(tanh(t1R * t2^T)); tanh bounds values so no softmax max-shift needed.
// The 64-column B panel is shared by all 8 waves of the block, so it is staged
// into LDS with double-buffered GLOBAL_LOAD_ASYNC_TO_LDS_B128 (ASYNCcnt):
// per 32-K chunk the panel is 64 rows x 64 B = 4 KiB = 256 threads x 16 B,
// i.e. exactly one async b128 instruction per wave per chunk.
__global__ void ke_gemm_A(const _Float16* __restrict__ A, const _Float16* __restrict__ Bm,
                          _Float16* __restrict__ expA) {
  __shared__ _Float16 panel[2][64 * 32];            // 2 x 4 KiB double buffer
  const int wave = threadIdx.x >> 5, lane = threadIdx.x & 31;
  const int hi = lane >> 4, ln = lane & 15;
  const int mrow = (blockIdx.y * 8 + wave) * 16;
  const int nbase = blockIdx.x * 64;
  v8f acc[4] = {};
  const _Float16* arow = A + (size_t)(mrow + ln) * 512;

  // per-thread 16B slice of the panel copy
  const int t = threadIdx.x;
  const int nloc = t >> 2;                          // 0..63 : panel row
  const int kbyte = (t & 3) * 16;                   // byte offset within 64B row
  const unsigned long long gbase =
      (unsigned long long)(uintptr_t)Bm + ((size_t)(nbase + nloc) * 512) * 2 + (unsigned)kbyte;
  // low 32 bits of the generic pointer == wave-relative LDS byte address
  const unsigned lds0 = (unsigned)(uintptr_t)&panel[0][0] + (unsigned)t * 16u;
  const unsigned lds1 = (unsigned)(uintptr_t)&panel[1][0] + (unsigned)t * 16u;

  asm volatile("global_load_async_to_lds_b128 %0, %1, off"
               :: "v"(lds0), "v"(gbase) : "memory");
  asm volatile("s_wait_asynccnt 0" ::: "memory");
  __syncthreads();

  for (int k0 = 0; k0 < 512; k0 += 32) {
    const int buf = (k0 >> 5) & 1;
    if (k0 + 32 < 512) {                            // prefetch next chunk into other buffer
      unsigned long long gn = gbase + (unsigned long long)(k0 + 32) * 2;
      asm volatile("global_load_async_to_lds_b128 %0, %1, off"
                   :: "v"(buf ? lds0 : lds1), "v"(gn) : "memory");
    }
    if (k0 + 64 < 512) __builtin_prefetch(arow + k0 + 64, 0, 0);  // global_prefetch_b8
    v16h a = load_frag(arow + k0, hi);
    #pragma unroll
    for (int nt = 0; nt < 4; ++nt) {
      v16h b = load_frag(&panel[buf][(nt * 16 + ln) * 32], hi);
      acc[nt] = wmma_f16(a, b, acc[nt]);
    }
    asm volatile("s_wait_asynccnt 0" ::: "memory");
    __syncthreads();                                 // next buffer landed; reads done
  }

  #pragma unroll
  for (int nt = 0; nt < 4; ++nt) {
    int n = nbase + nt * 16 + ln;
    #pragma unroll
    for (int r = 0; r < 8; ++r) {
      int row = mrow + hi * 8 + r;
      expA[(size_t)row * 4096 + n] = (_Float16)__expf(tanhf(acc[nt][r]));
    }
  }
}

__global__ void ke_rowsum(const _Float16* __restrict__ expA, float* __restrict__ rs) {
  __shared__ float red[256];
  const _Float16* rp = expA + (size_t)blockIdx.x * 4096;
  float s = 0.f;
  for (int j = threadIdx.x; j < 4096; j += 256) s += (float)rp[j];
  red[threadIdx.x] = s; __syncthreads();
  for (int st = 128; st > 0; st >>= 1) {
    if (threadIdx.x < st) red[threadIdx.x] += red[threadIdx.x + st];
    __syncthreads();
  }
  if (threadIdx.x == 0) rs[blockIdx.x] = red[0];
}

__global__ void ke_colsum(const _Float16* __restrict__ expA, float* __restrict__ cs) {
  int j = blockIdx.x * 256 + threadIdx.x;          // 4096 columns
  float s = 0.f;
  for (int i = 0; i < 4096; ++i) s += (float)expA[(size_t)i * 4096 + j];
  cs[j] = s;
}

__global__ void ke_sf1(const _Float16* __restrict__ expA, const float* __restrict__ rs,
                       float* __restrict__ sf1) {
  int j = blockIdx.x * 256 + threadIdx.x;
  float s = 0.f;
  for (int i = 0; i < 4096; ++i) s += (float)expA[(size_t)i * 4096 + j] / rs[i];
  sf1[j] = s * (1.f / 4096.f);
}

__global__ void ke_sf2(const _Float16* __restrict__ expA, const float* __restrict__ cs,
                       float* __restrict__ sf2) {
  __shared__ float red[256];
  const _Float16* rp = expA + (size_t)blockIdx.x * 4096;
  float s = 0.f;
  for (int j = threadIdx.x; j < 4096; j += 256) s += (float)rp[j] / cs[j];
  red[threadIdx.x] = s; __syncthreads();
  for (int st = 128; st > 0; st >>= 1) {
    if (threadIdx.x < st) red[threadIdx.x] += red[threadIdx.x + st];
    __syncthreads();
  }
  if (threadIdx.x == 0) sf2[blockIdx.x] = red[0] * (1.f / 4096.f);
}

// out[b, 0:128, :] = t1*sf2 ; out[b, 128:256, :] = t2*sf1 (flattened to 131072 cols)
__global__ void ke_out(const float* __restrict__ r32, const float* __restrict__ sf1,
                       const float* __restrict__ sf2, float* __restrict__ outb) {
  int idx = blockIdx.x * 256 + threadIdx.x;        // 32*131072
  int b = idx >> 17, rem = idx & (OUTC - 1);
  int l2 = rem >> 9, h2 = rem & 511;
  float v;
  if (l2 < 128) {
    int row = b * 128 + l2;
    v = r32[(size_t)row * 512 + h2] * sf2[row];
  } else {
    int row = b * 128 + (l2 - 128);
    v = r32[(size_t)4096 * 512 + (size_t)row * 512 + h2] * sf1[row];
  }
  outb[idx] = v;
}

__global__ void ke_bn(const float* __restrict__ outb, float* __restrict__ mu,
                      float* __restrict__ rstd) {
  int c = blockIdx.x * 256 + threadIdx.x;          // 131072 columns
  float s = 0.f, s2 = 0.f;
  for (int b = 0; b < 32; ++b) {
    float v = outb[(size_t)b * OUTC + c];
    s += v; s2 += v * v;
  }
  float m = s * (1.f / 32.f);
  mu[c] = m;
  rstd[c] = rsqrtf(s2 * (1.f / 32.f) - m * m + 1e-5f);
}

__global__ void ke_fc(const float* __restrict__ outb, const float* __restrict__ mu,
                      const float* __restrict__ rstd, const float* __restrict__ gamma,
                      const float* __restrict__ beta, const float* __restrict__ fcw,
                      const float* __restrict__ fcb, float* __restrict__ dout) {
  __shared__ float red[4][256];
  int b = blockIdx.x, tid = threadIdx.x;
  float a0 = 0, a1 = 0, a2 = 0, a3 = 0;
  const float* ob = outb + (size_t)b * OUTC;
  for (int col = tid; col < OUTC; col += 256) {
    float v = (ob[col] - mu[col]) * rstd[col] * gamma[col] + beta[col];
    a0 += v * fcw[col];
    a1 += v * fcw[OUTC + col];
    a2 += v * fcw[2 * OUTC + col];
    a3 += v * fcw[3 * OUTC + col];
  }
  red[0][tid] = a0; red[1][tid] = a1; red[2][tid] = a2; red[3][tid] = a3;
  __syncthreads();
  for (int st = 128; st > 0; st >>= 1) {
    if (tid < st) {
      red[0][tid] += red[0][tid + st]; red[1][tid] += red[1][tid + st];
      red[2][tid] += red[2][tid + st]; red[3][tid] += red[3][tid + st];
    }
    __syncthreads();
  }
  if (tid == 0) {
    float l[4];
    #pragma unroll
    for (int cc = 0; cc < 4; ++cc) l[cc] = red[cc][0] + fcb[cc];
    float m = fmaxf(fmaxf(l[0], l[1]), fmaxf(l[2], l[3]));
    float s = __expf(l[0] - m) + __expf(l[1] - m) + __expf(l[2] - m) + __expf(l[3] - m);
    float lse = m + __logf(s);
    #pragma unroll
    for (int cc = 0; cc < 4; ++cc) dout[b * 4 + cc] = l[cc] - lse;
  }
}

// ---------------- launcher ----------------
extern "C" void kernel_launch(void* const* d_in, const int* in_sizes, int n_in,
                              void* d_out, int out_size, void* d_ws, size_t ws_size,
                              hipStream_t stream) {
  const int*   x     = (const int*)  d_in[0];
  const float* tbl   = (const float*)d_in[2];
  const float* RM    = (const float*)d_in[3];
  const float* wihf  = (const float*)d_in[4];
  const float* whhf  = (const float*)d_in[5];
  const float* bf    = (const float*)d_in[6];
  const float* wihb  = (const float*)d_in[7];
  const float* whhb  = (const float*)d_in[8];
  const float* bb    = (const float*)d_in[9];
  const float* gamma = (const float*)d_in[10];
  const float* beta  = (const float*)d_in[11];
  const float* fcw   = (const float*)d_in[12];
  const float* fcb   = (const float*)d_in[13];
  (void)in_sizes; (void)n_in; (void)out_size; (void)ws_size;

  char* ws = (char*)d_ws;
  _Float16* e16   = (_Float16*)(ws + OFF_E16);
  _Float16* wih16 = (_Float16*)(ws + OFF_WIH16);
  _Float16* whh16 = (_Float16*)(ws + OFF_WHH16);
  _Float16* rmT16 = (_Float16*)(ws + OFF_RMT16);
  _Float16* xw16  = (_Float16*)(ws + OFF_XW16);
  float*    r32   = (float*)   (ws + OFF_R32);
  _Float16* r16   = (_Float16*)(ws + OFF_R16);
  _Float16* t1R16 = (_Float16*)(ws + OFF_T1R);
  _Float16* expA  = (_Float16*)(ws + OFF_EXPA);
  float*    rs    = (float*)   (ws + OFF_RS);
  float*    cs    = (float*)   (ws + OFF_CS);
  float*    sf1   = (float*)   (ws + OFF_SF1);
  float*    sf2   = (float*)   (ws + OFF_SF2);
  float*    outb  = (float*)   (ws + OFF_OUT);
  float*    mu    = (float*)   (ws + OFF_MU);
  float*    rstd  = (float*)   (ws + OFF_RSTD);

  ke_embed<<<8192, 256, 0, stream>>>(x, tbl, e16);
  ke_cvt  <<<1024, 256, 0, stream>>>(wihf, wih16,           1024 * 256);
  ke_cvt  <<<1024, 256, 0, stream>>>(wihb, wih16 + 262144,  1024 * 256);
  ke_cvt  <<<1024, 256, 0, stream>>>(whhf, whh16,           1024 * 256);
  ke_cvt  <<<1024, 256, 0, stream>>>(whhb, whh16 + 262144,  1024 * 256);
  ke_cvtT <<<1024, 256, 0, stream>>>(RM, rmT16);

  ke_gemm_xw<<<dim3(16, 64, 2), 256, 0, stream>>>(e16, wih16, bf, bb, xw16);
  ke_lstm   <<<4, 512, 0, stream>>>(xw16, whh16, r32, r16);

  ke_gemm_t1R<<<dim3(8, 32), 256, 0, stream>>>(r16, rmT16, t1R16);
  ke_gemm_A  <<<dim3(64, 32), 256, 0, stream>>>(t1R16, r16 + (size_t)4096 * 512, expA);

  ke_rowsum<<<4096, 256, 0, stream>>>(expA, rs);
  ke_colsum<<<16,   256, 0, stream>>>(expA, cs);
  ke_sf1   <<<16,   256, 0, stream>>>(expA, rs, sf1);
  ke_sf2   <<<4096, 256, 0, stream>>>(expA, cs, sf2);

  ke_out<<<16384, 256, 0, stream>>>(r32, sf1, sf2, outb);
  ke_bn <<<512,   256, 0, stream>>>(outb, mu, rstd);
  ke_fc <<<32,    256, 0, stream>>>(outb, mu, rstd, gamma, beta, fcw, fcb, (float*)d_out);
}
